// Differential_21586505630085
// MI455X (gfx1250) — matrix operand order repaired and checked by reading, not compiled, not run on hardware
//
#include <hip/hip_runtime.h>
#include <math.h>

#define B_  2
#define H_  16
#define S_  2048
#define D_  128
#define LAMBDA_INIT 0.8f
#define GN_EPS 1e-5f
#define BM 64            // query rows per block
#define BN 64            // key tile
#define NTHREADS 256     // 8 wave32s

typedef __attribute__((ext_vector_type(16))) _Float16 v16h;
typedef __attribute__((ext_vector_type(4)))  _Float16 v4h;
typedef __attribute__((ext_vector_type(8)))  float    v8f;
typedef __attribute__((ext_vector_type(4)))  float    v4f;

__device__ __forceinline__ v8f wmma16x16x32(v16h a, v16h b, v8f c) {
  // D = A(16x32 f16) * B(32x16 f16) + C(16x16 f32)
  return __builtin_amdgcn_wmma_f32_16x16x32_f16(false, a, false, b, (short)0, c,
                                                false, false);
}

// ---------------------------------------------------------------------------
// Kernel 0: lambda scalar + zero the per-(b,h) sum / sumsq accumulators.
// ws[0] = lambda; ws[1 + 2*bh] = sum; ws[2 + 2*bh] = sumsq   (bh in [0,32))
// ---------------------------------------------------------------------------
__global__ void diffattn_init_kernel(const float* __restrict__ lq1,
                                     const float* __restrict__ lq2,
                                     const float* __restrict__ lk1,
                                     const float* __restrict__ lk2,
                                     float* __restrict__ ws) {
  __shared__ float s1[128];
  __shared__ float s2[128];
  int t = threadIdx.x;
  s1[t] = lq1[t] * lk1[t];
  s2[t] = lq2[t] * lk2[t];
  __syncthreads();
  for (int o = 64; o > 0; o >>= 1) {
    if (t < o) { s1[t] += s1[t + o]; s2[t] += s2[t + o]; }
    __syncthreads();
  }
  if (t == 0) ws[0] = __expf(s1[0]) - __expf(s2[0]) + LAMBDA_INIT;
  if (t >= 1 && t <= 2 * B_ * H_) ws[t] = 0.0f;
}

// ---------------------------------------------------------------------------
// Kernel 1: fused differential flash attention.
// Grid: B*H*(S/BM) blocks of 256 threads. Waves 0-3 -> stream 1 (Q1K1),
// waves 4-7 -> stream 2 (Q2K2); each wave owns 16 query rows.
// Writes UNNORMALIZED (w1 - lam*w2) @ v to d_out, accumulates stats in ws.
// ---------------------------------------------------------------------------
__global__ void __launch_bounds__(NTHREADS)
diffattn_fa_kernel(const float* __restrict__ q,
                   const float* __restrict__ k,
                   const float* __restrict__ v,
                   float* __restrict__ out,
                   float* __restrict__ ws) {
  // 64KB static LDS, overlaid:
  //   phase A: Kbuf f16 [2][64][128] (32KB) | Vt f16 [128][64] (16KB) | Pbuf f16 [8][16][64] (16KB)
  //   phase B: comb f32 [2][64][128] (64KB)
  __shared__ __align__(32) unsigned char smem[64 * 1024];
  _Float16* Kbuf = (_Float16*)smem;                    // [s][key][feat]
  _Float16* Vt   = (_Float16*)(smem + 32 * 1024);      // [feat][key] (transposed)
  _Float16* Pbuf = (_Float16*)(smem + 48 * 1024);      // [wave][row][key]
  float*    comb = (float*)smem;                       // [s][row][feat]

  const int tid    = threadIdx.x;
  const int lane   = tid & 31;
  const int wave   = tid >> 5;
  const int rg     = wave & 3;   // row-group 0..3 (16 rows each)
  const int stream = wave >> 2;  // 0 => Q1K1, 1 => Q2K2
  const int mlane  = lane & 15;  // M (or N) index within tile
  const int half   = lane >> 4;  // lane half selects K sub-range

  const int nqt  = S_ / BM;
  const int bh   = blockIdx.x / nqt;
  const int qt   = blockIdx.x % nqt;
  const int q0   = qt * BM;

  const float scale = 0.08838834764831845f; // 1/sqrt(128), folded into Q

  // ---- Q A-fragments: 16 rows x 128 feat, scale pre-folded, in registers ----
  // A 16x32 layout (contiguous, matching B): lane(m,half) holds K = kb+half*16+{0..15}
  // in elements [0..15].
  const float* qg = q + ((size_t)bh * S_ + (q0 + rg * 16 + mlane)) * (2 * D_)
                      + stream * D_;
  v16h aq[4];
  #pragma unroll
  for (int f = 0; f < 4; ++f) {
    const int kb = f * 32;
    const float* c0 = qg + kb + half * 16;
    const float* c1 = qg + kb + half * 16 + 8;
    v4f a0 = *(const v4f*)(c0);     v4f a1 = *(const v4f*)(c0 + 4);
    v4f b0 = *(const v4f*)(c1);     v4f b1 = *(const v4f*)(c1 + 4);
    v16h t;
    #pragma unroll
    for (int e = 0; e < 4; ++e) {
      t[e]      = (_Float16)(a0[e] * scale);  t[4 + e]  = (_Float16)(a1[e] * scale);
      t[8 + e]  = (_Float16)(b0[e] * scale);  t[12 + e] = (_Float16)(b1[e] * scale);
    }
    aq[f] = t;
  }

  // ---- online-softmax state: row = cc + 8*half within this wave's 16 rows ----
  float m_run[8], l_run[8];
  v8f   o_acc[8];
  const v8f vzero = {0.f, 0.f, 0.f, 0.f, 0.f, 0.f, 0.f, 0.f};
  #pragma unroll
  for (int cc = 0; cc < 8; ++cc) { m_run[cc] = -INFINITY; l_run[cc] = 0.0f; }
  #pragma unroll
  for (int nt = 0; nt < 8; ++nt) o_acc[nt] = vzero;

  const float* kg = k + (size_t)bh * S_ * (2 * D_);
  const float* vg = v + (size_t)bh * S_ * D_;
  _Float16* pmy = Pbuf + wave * (16 * 64);

  #pragma unroll 1
  for (int kt = 0; kt < S_; kt += BN) {
    __syncthreads();
    // -- stage K tile (both streams) as f16, vectorized b128 -> b64 --
    for (int idx = tid; idx < (2 * BN * D_) / 4; idx += NTHREADS) {
      int e4  = idx * 4;
      int s2  = e4 >> 13;          // /(64*128)
      int rem = e4 & 8191;
      int r = rem >> 7, f = rem & 127;
      v4f src = *(const v4f*)(kg + (size_t)(kt + r) * (2 * D_) + s2 * D_ + f);
      v4h d;
      #pragma unroll
      for (int e = 0; e < 4; ++e) d[e] = (_Float16)src[e];
      *(v4h*)(Kbuf + e4) = d;
    }
    // -- stage V tile transposed: Vt[f][r], vectorized global reads --
    for (int idx = tid; idx < (BN * D_) / 4; idx += NTHREADS) {
      int e4 = idx * 4;
      int r = e4 >> 7, f = e4 & 127;
      v4f src = *(const v4f*)(vg + (size_t)(kt + r) * D_ + f);
      #pragma unroll
      for (int e = 0; e < 4; ++e) Vt[(f + e) * BN + r] = (_Float16)src[e];
    }
    __syncthreads();

    // -- prefetch next key tile into cache while we compute this one --
    if (kt + BN < S_) {
      __builtin_prefetch(kg + (size_t)(kt + BN) * (2 * D_) + (size_t)tid * 64, 0, 0);
      __builtin_prefetch(vg + (size_t)(kt + BN) * D_ + (size_t)tid * 32, 0, 0);
    }

    // -- scores: 4 N-tiles of 16 keys, K-loop over D in steps of 32 --
    v8f st[4];
    #pragma unroll
    for (int j = 0; j < 4; ++j) {
      v8f c = vzero;
      #pragma unroll
      for (int f = 0; f < 4; ++f) {
        // B 32x16 f16 layout: lane(n,half) holds K=kb+half*16+{0..15} of key n
        const _Float16* kr = Kbuf + stream * 8192 + (j * 16 + mlane) * D_
                                  + f * 32 + half * 16;
        v16h bq = *(const v16h*)kr;
        c = wmma16x16x32(aq[f], bq, c);
      }
      st[j] = c;  // scale already folded into Q
    }

    // -- row max across the 16 keys-per-tile (lanes) and 4 tiles --
    float corr[8], rs[8];
    #pragma unroll
    for (int cc = 0; cc < 8; ++cc) {
      float mx = st[0][cc];
      #pragma unroll
      for (int j = 1; j < 4; ++j) mx = fmaxf(mx, st[j][cc]);
      #pragma unroll
      for (int off = 1; off < 16; off <<= 1)
        mx = fmaxf(mx, __shfl_xor(mx, off, 16));
      float mn = fmaxf(m_run[cc], mx);
      corr[cc] = __expf(m_run[cc] - mn);
      m_run[cc] = mn;
      rs[cc] = 0.0f;
    }
    // -- exponentiate, row sums --
    #pragma unroll
    for (int j = 0; j < 4; ++j)
      #pragma unroll
      for (int cc = 0; cc < 8; ++cc) {
        float p = __expf(st[j][cc] - m_run[cc]);
        st[j][cc] = p;
        rs[cc] += p;
      }
    #pragma unroll
    for (int cc = 0; cc < 8; ++cc) {
      float r = rs[cc];
      #pragma unroll
      for (int off = 1; off < 16; off <<= 1) r += __shfl_xor(r, off, 16);
      l_run[cc] = l_run[cc] * corr[cc] + r;
    }
    // -- rescale O accumulators --
    #pragma unroll
    for (int nt = 0; nt < 8; ++nt)
      #pragma unroll
      for (int cc = 0; cc < 8; ++cc) o_acc[nt][cc] *= corr[cc];

    // -- C-layout P -> LDS (f16) -> A-layout fragments (wave-private region) --
    #pragma unroll
    for (int j = 0; j < 4; ++j)
      #pragma unroll
      for (int cc = 0; cc < 8; ++cc)
        pmy[(cc + 8 * half) * BN + j * 16 + mlane] = (_Float16)st[j][cc];

    v16h ap[2];
    #pragma unroll
    for (int f = 0; f < 2; ++f) {
      const _Float16* pr = pmy + mlane * BN + f * 32 + half * 16;
      v16h t;
      #pragma unroll
      for (int e = 0; e < 8; ++e) { t[e] = pr[e]; t[8 + e] = pr[8 + e]; }
      ap[f] = t;
    }

    // -- O += P @ V : 8 feature N-tiles, K-loop over 64 keys in steps of 32 --
    #pragma unroll
    for (int nt = 0; nt < 8; ++nt)
      #pragma unroll
      for (int f = 0; f < 2; ++f) {
        const _Float16* vr = Vt + (nt * 16 + mlane) * BN + f * 32 + half * 16;
        v16h bv = *(const v16h*)vr;
        o_acc[nt] = wmma16x16x32(ap[f], bv, o_acc[nt]);
      }
  }

  // ---- finalize: O / l, combine streams via LDS overlay ----
  float inv_l[8];
  #pragma unroll
  for (int cc = 0; cc < 8; ++cc) inv_l[cc] = 1.0f / l_run[cc];

  __syncthreads();  // done with Kbuf/Vt/Pbuf -> reuse as comb
  #pragma unroll
  for (int nt = 0; nt < 8; ++nt)
    #pragma unroll
    for (int cc = 0; cc < 8; ++cc)
      comb[stream * 8192 + (rg * 16 + cc + 8 * half) * D_ + nt * 16 + mlane] =
          o_acc[nt][cc] * inv_l[cc];
  __syncthreads();

  const float lam = ws[0];
  float lsum = 0.0f, lss = 0.0f;
  float* orow = out + ((size_t)bh * S_ + q0) * D_;
  for (int idx = tid; idx < BM * D_; idx += NTHREADS) {
    float val = comb[idx] - lam * comb[8192 + idx];
    orow[idx] = val;
    lsum += val;
    lss  += val * val;
  }
  // wave reduce + one atomic per wave into the per-(b,h) stats
  #pragma unroll
  for (int off = 1; off < 32; off <<= 1) {
    lsum += __shfl_xor(lsum, off, 32);
    lss  += __shfl_xor(lss,  off, 32);
  }
  if (lane == 0) {
    atomicAdd(&ws[1 + 2 * bh], lsum);
    atomicAdd(&ws[2 + 2 * bh], lss);
  }
}

// ---------------------------------------------------------------------------
// Kernel 2: GroupNorm over each (b,h) slab of S*D, in place on d_out,
// then gamma/beta and the (1 - LAMBDA_INIT) factor.
// ---------------------------------------------------------------------------
__global__ void diffattn_gn_kernel(float* __restrict__ out,
                                   const float* __restrict__ gw,
                                   const float* __restrict__ gb,
                                   const float* __restrict__ ws) {
  size_t i = (size_t)blockIdx.x * NTHREADS + threadIdx.x;
  const size_t total = (size_t)B_ * H_ * S_ * D_;
  if (i >= total) return;
  const int bh = (int)(i / ((size_t)S_ * D_));
  const int h  = bh % H_;
  const int f  = (int)(i % D_);
  const float n    = (float)(S_ * D_);
  const float sum  = ws[1 + 2 * bh];
  const float ss   = ws[2 + 2 * bh];
  const float mean = sum / n;
  const float var  = ss / n - mean * mean;
  const float x = out[i];
  const float g = gw[h * D_ + f];
  const float b = gb[h * D_ + f];
  out[i] = ((x - mean) * rsqrtf(var + GN_EPS) * g + b) * (1.0f - LAMBDA_INIT);
}

// ---------------------------------------------------------------------------
extern "C" void kernel_launch(void* const* d_in, const int* in_sizes, int n_in,
                              void* d_out, int out_size, void* d_ws, size_t ws_size,
                              hipStream_t stream) {
  const float* q   = (const float*)d_in[0];
  const float* k   = (const float*)d_in[1];
  const float* v   = (const float*)d_in[2];
  const float* lq1 = (const float*)d_in[3];
  const float* lq2 = (const float*)d_in[4];
  const float* lk1 = (const float*)d_in[5];
  const float* lk2 = (const float*)d_in[6];
  const float* gw  = (const float*)d_in[7];
  const float* gb  = (const float*)d_in[8];
  float* out = (float*)d_out;
  float* ws  = (float*)d_ws;

  diffattn_init_kernel<<<1, 128, 0, stream>>>(lq1, lq2, lk1, lk2, ws);

  const int nblocks = B_ * H_ * (S_ / BM);  // 1024
  diffattn_fa_kernel<<<nblocks, NTHREADS, 0, stream>>>(q, k, v, out, ws);

  const size_t total = (size_t)B_ * H_ * S_ * D_;
  const int gnblocks = (int)((total + NTHREADS - 1) / NTHREADS);
  diffattn_gn_kernel<<<gnblocks, NTHREADS, 0, stream>>>(out, gw, gb, ws);
}
